// PhaseToRatePipeline_61435212202268
// MI455X (gfx1250) — matrix operand order repaired and checked by reading, not compiled, not run on hardware
//
#include <hip/hip_runtime.h>

// ---------------------------------------------------------------------------
// PhaseToRatePipeline on MI455X (gfx1250, wave32, WMMA + TDM)
//   K1: Kuramoto sim + exact per-row quantile + soft gate -> f16 rate matrix
//   K2: W (f32 1000x704) -> f16, zero-padded to 1024 rows
//   K3: rate(65536x704) x W^T + b via v_wmma_f32_16x16x32_f16, with the
//       per-block 64x704 W strip DMA'd into LDS by the Tensor Data Mover.
// ---------------------------------------------------------------------------

#define N_DELTA   64
#define N_THETA   128
#define N_GAMMA   512
#define N_TOTAL   704          // = 22 * 32 lane-chunks
#define N_CHUNK   22
#define N_CLASSES 1000
#define N_PAD     1024         // padded class dim (16 strips of 64)
#define B_ROWS    65536
#define DT_C      0.01f
#define TWO_PI_F  6.283185307179586f
#define COUPLING  2.0f
#define PAC       0.3f

typedef _Float16 v16h __attribute__((ext_vector_type(16)));
typedef _Float16 v8h  __attribute__((ext_vector_type(8)));
typedef float    v8f  __attribute__((ext_vector_type(8)));
typedef unsigned int u32x4 __attribute__((ext_vector_type(4)));
typedef int          i32x4 __attribute__((ext_vector_type(4)));
typedef int          i32x8 __attribute__((ext_vector_type(8)));

__device__ __forceinline__ float wave_sumf(float v) {
#pragma unroll
    for (int o = 16; o > 0; o >>= 1) v += __shfl_xor(v, o, 32);
    return v;
}
__device__ __forceinline__ int wave_sumi(int v) {
#pragma unroll
    for (int o = 16; o > 0; o >>= 1) v += __shfl_xor(v, o, 32);
    return v;
}
__device__ __forceinline__ unsigned wave_minu(unsigned v) {
#pragma unroll
    for (int o = 16; o > 0; o >>= 1) {
        unsigned t = (unsigned)__shfl_xor((int)v, o, 32);
        v = (t < v) ? t : v;
    }
    return v;
}

// ---------------------------------------------------------------------------
// Kernel 1: oscillator simulation, one wave per sample row.
// Element e = chunk*32 + lane; band boundaries (64,192) are chunk-aligned.
// ---------------------------------------------------------------------------
__global__ __launch_bounds__(256) void sim_kernel(
    const float* __restrict__ x,
    const float* __restrict__ phase0,
    _Float16*    __restrict__ rateH)
{
    const int wave = threadIdx.x >> 5;
    const int lane = threadIdx.x & 31;
    const int row  = blockIdx.x * 8 + wave;

    const float* xr = x      + (size_t)row * N_TOTAL;
    const float* pr = phase0 + (size_t)row * N_TOTAL;

    float phi[N_CHUNK], amp[N_CHUNK], w[N_CHUNK];
#pragma unroll
    for (int j = 0; j < N_CHUNK; ++j) {
        const int e = j * 32 + lane;
        phi[j] = pr[e] * TWO_PI_F;
        amp[j] = 1.0f + fabsf(xr[e]) * 0.01f;
        float f;
        if (e < N_DELTA)                 f = 1.0f  +  3.0f * (float)e         * (1.0f / 63.0f);
        else if (e < N_DELTA + N_THETA)  f = 4.0f  +  4.0f * (float)(e - 64)  * (1.0f / 127.0f);
        else                             f = 30.0f + 50.0f * (float)(e - 192) * (1.0f / 511.0f);
        w[j] = TWO_PI_F * f;
    }

    // Delta amps fixed; theta/gamma amps share one multiplicative factor/step.
    float multT = 1.0f, multG = 1.0f;

#pragma unroll
    for (int step = 0; step < 5; ++step) {
        float s[N_CHUNK], c[N_CHUNK];
        float Sd = 0.f, Cd = 0.f, St = 0.f, Ct = 0.f, Sg = 0.f, Cg = 0.f;
#pragma unroll
        for (int j = 0; j < N_CHUNK; ++j) {
            __sincosf(phi[j], &s[j], &c[j]);
            if (j < 2)      { Sd += s[j]; Cd += c[j]; }
            else if (j < 6) { St += s[j]; Ct += c[j]; }
            else            { Sg += s[j]; Cg += c[j]; }
        }
        Sd = wave_sumf(Sd); Cd = wave_sumf(Cd);
        St = wave_sumf(St); Ct = wave_sumf(Ct);
        Sg = wave_sumf(Sg); Cg = wave_sumf(Cg);

        // cos(atan2(S,C)) = C / sqrt(S^2 + C^2)
        multT *= 1.0f + DT_C * PAC * (Cd * __frsqrt_rn(Sd * Sd + Cd * Cd + 1e-30f));
        multG *= 1.0f + DT_C * PAC * (Ct * __frsqrt_rn(St * St + Ct * Ct + 1e-30f));

        const float smD = Sd * (1.0f / 64.0f),  cmD = Cd * (1.0f / 64.0f);
        const float smT = St * (1.0f / 128.0f), cmT = Ct * (1.0f / 128.0f);
        const float smG = Sg * (1.0f / 512.0f), cmG = Cg * (1.0f / 512.0f);
#pragma unroll
        for (int j = 0; j < N_CHUNK; ++j) {
            float sm, cm;
            if (j < 2)      { sm = smD; cm = cmD; }
            else if (j < 6) { sm = smT; cm = cmT; }
            else            { sm = smG; cm = cmG; }
            const float dphi = w[j] + COUPLING * (sm * c[j] - cm * s[j]);
            phi[j] = fmodf(phi[j] + DT_C * dphi, TWO_PI_F);
        }
    }

    float raw[N_CHUNK];
#pragma unroll
    for (int j = 0; j < N_CHUNK; ++j) {
        const float m = (j < 2) ? 1.0f : (j < 6) ? multT : multG;
        raw[j] = amp[j] * m * 0.5f * (1.0f + __cosf(phi[j]));
    }

    // Exact order stats s[632], s[633] via binary search on IEEE bits (raw>=0).
    unsigned key[N_CHUNK];
#pragma unroll
    for (int j = 0; j < N_CHUNK; ++j) key[j] = __float_as_uint(raw[j]);

    unsigned lo = 0u, hi = 0x7F800000u;
    for (int it = 0; it < 31; ++it) {
        const unsigned mid = (lo + hi) >> 1;
        int cnt = 0;
#pragma unroll
        for (int j = 0; j < N_CHUNK; ++j) cnt += (key[j] <= mid) ? 1 : 0;
        cnt = wave_sumi(cnt);
        if (cnt >= 633) hi = mid; else lo = mid + 1;
    }
    const unsigned k632 = hi;
    int ceq = 0;
    unsigned above = 0x7F800000u;
#pragma unroll
    for (int j = 0; j < N_CHUNK; ++j) {
        if (key[j] <= k632) ceq++;
        else if (key[j] < above) above = key[j];
    }
    ceq   = wave_sumi(ceq);
    above = wave_minu(above);
    const unsigned k633 = (ceq >= 634) ? k632 : above;
    const float thr = 0.3f * __uint_as_float(k632) + 0.7f * __uint_as_float(k633);

    _Float16* outp = rateH + (size_t)row * N_TOTAL;
#pragma unroll
    for (int j = 0; j < N_CHUNK; ++j) {
        const float g = raw[j] / (1.0f + __expf(-10.0f * (raw[j] - thr)));
        outp[j * 32 + lane] = (_Float16)g;
    }
}

// ---------------------------------------------------------------------------
// Kernel 2: W f32 -> f16, zero-padded to N_PAD rows.
// ---------------------------------------------------------------------------
__global__ __launch_bounds__(256) void convw_kernel(
    const float* __restrict__ W, _Float16* __restrict__ Wh)
{
    const int idx = blockIdx.x * 256 + threadIdx.x;   // < N_PAD * N_TOTAL
    const int r = idx / N_TOTAL;
    const int k = idx - r * N_TOTAL;
    Wh[idx] = (_Float16)((r < N_CLASSES) ? W[(size_t)r * N_TOTAL + k] : 0.0f);
}

// ---------------------------------------------------------------------------
// Kernel 3: out = rate @ W^T + b.
// Block = 8 waves; block owns a 64-col W strip (64x704 f16, 90112 B) staged
// in LDS by one TENSOR_LOAD_TO_LDS (TDM). Each wave computes 32 rows x 64
// cols: 8 accumulators, A fragments reused x4, B fragments reused x2 and
// shared block-wide through LDS.
// Fragment layouts (ISA 7.12.2):
//   A: lane l -> row m+l%16, K in {k0..k0+7} U {k0+16..k0+23}, k0=8*(l>=16)
//   B: lane l -> col n+l%16, K in {k0..k0+7} U {k0+16..k0+23}, k0=8*(l>=16) (same k order as A)
// ---------------------------------------------------------------------------
__global__ __launch_bounds__(256) void gemm_kernel(
    const _Float16* __restrict__ A,     // rate, B_ROWS x N_TOTAL
    const _Float16* __restrict__ Bm,    // Wh,   N_PAD  x N_TOTAL
    const float*    __restrict__ bias,  // N_CLASSES
    float*          __restrict__ out)   // B_ROWS x N_CLASSES
{
    extern __shared__ _Float16 ldsB[];  // 64 * 704 halves = 90112 bytes

    const int wave = threadIdx.x >> 5;
    const int lane = threadIdx.x & 31;
    const int ln   = lane & 15;
    const int lh   = lane >> 4;
    const int m0   = (blockIdx.x * 8 + wave) * 32;   // 2 M-tiles per wave
    const int n0   = blockIdx.y * 64;

    // ---- TDM: DMA this block's 64x704 f16 W strip into LDS (wave 0) ----
    if (wave == 0) {
        const unsigned lds_off = __builtin_amdgcn_groupstaticsize(); // dyn LDS start
        const unsigned long long ga =
            (unsigned long long)(size_t)(Bm + (size_t)n0 * N_TOTAL);
        u32x4 g0;
        g0[0] = 1u;                                             // count=1, user desc
        g0[1] = lds_off;                                        // lds_addr (bytes)
        g0[2] = (unsigned)ga;                                   // global_addr[31:0]
        g0[3] = (unsigned)((ga >> 32) & 0x01FFFFFFu) | 0x80000000u; // [56:32] | type=2
        i32x8 g1;
        g1[0] = (int)(1u << 16);                  // wg_mask=0, data_size=2B
        g1[1] = (int)((unsigned)N_TOTAL << 16);   // tensor_dim0 lo16 (704)
        g1[2] = (int)((unsigned)N_PAD << 16);     // tensor_dim0 hi=0 | tensor_dim1 lo16
        g1[3] = (int)((unsigned)N_TOTAL << 16);   // tensor_dim1 hi=0 | tile_dim0=704
        g1[4] = 64;                               // tile_dim1=64, tile_dim2=0
        g1[5] = N_TOTAL;                          // tensor_dim0_stride lo32 (elements)
        g1[6] = 0;                                // stride hi | tensor_dim1_stride lo
        g1[7] = 0;
        const i32x4 z4 = {0, 0, 0, 0};
        const i32x8 z8 = {0, 0, 0, 0, 0, 0, 0, 0};
        __builtin_amdgcn_tensor_load_to_lds(g0, g1, z4, z4, z8, 0);
        __builtin_amdgcn_s_wait_tensorcnt(0);
    }
    __syncthreads();

    v8f acc0[4] = {}, acc1[4] = {};

    const _Float16* arow0 = A + (size_t)(m0 + ln) * N_TOTAL + lh * 8;
    const _Float16* arow1 = arow0 + (size_t)16 * N_TOTAL;
    const _Float16* lrow  = ldsB + (size_t)ln * N_TOTAL + lh * 8;

    for (int t = 0; t < N_CHUNK; ++t) {
        const v8h a0lo = *(const v8h*)(arow0 + t * 32);
        const v8h a0hi = *(const v8h*)(arow0 + t * 32 + 16);
        const v8h a1lo = *(const v8h*)(arow1 + t * 32);
        const v8h a1hi = *(const v8h*)(arow1 + t * 32 + 16);
        const v16h a0 = __builtin_shufflevector(a0lo, a0hi,
            0, 1, 2, 3, 4, 5, 6, 7, 8, 9, 10, 11, 12, 13, 14, 15);
        const v16h a1 = __builtin_shufflevector(a1lo, a1hi,
            0, 1, 2, 3, 4, 5, 6, 7, 8, 9, 10, 11, 12, 13, 14, 15);

#pragma unroll
        for (int j = 0; j < 4; ++j) {
            const _Float16* bp = lrow + (size_t)j * 16 * N_TOTAL + t * 32;
            const v8h blo = *(const v8h*)(bp);
            const v8h bhi = *(const v8h*)(bp + 16);
            const v16h b = __builtin_shufflevector(blo, bhi,
                0, 1, 2, 3, 4, 5, 6, 7, 8, 9, 10, 11, 12, 13, 14, 15);
            acc0[j] = __builtin_amdgcn_wmma_f32_16x16x32_f16(false, a0, false, b, (short)0, acc0[j], false, false);
            acc1[j] = __builtin_amdgcn_wmma_f32_16x16x32_f16(false, a1, false, b, (short)0, acc1[j], false, false);
        }
    }

    // D layout: VGPR r -> M = m + r + 8*(lane>=16), N = n0 + 16*j + lane%16
#pragma unroll
    for (int j = 0; j < 4; ++j) {
        const int n = n0 + j * 16 + ln;
        if (n < N_CLASSES) {
            const float bv = bias[n];
#pragma unroll
            for (int r = 0; r < 8; ++r) {
                const int mr = r + lh * 8;
                out[(size_t)(m0 + mr) * N_CLASSES + n]      = acc0[j][r] + bv;
                out[(size_t)(m0 + 16 + mr) * N_CLASSES + n] = acc1[j][r] + bv;
            }
        }
    }
}

// ---------------------------------------------------------------------------
extern "C" void kernel_launch(void* const* d_in, const int* in_sizes, int n_in,
                              void* d_out, int out_size, void* d_ws, size_t ws_size,
                              hipStream_t stream)
{
    (void)in_sizes; (void)n_in; (void)out_size; (void)ws_size;

    const float* x      = (const float*)d_in[0];   // (B, 704)
    const float* phase0 = (const float*)d_in[1];   // (B, 704)
    const float* W      = (const float*)d_in[2];   // (1000, 704)
    const float* bias   = (const float*)d_in[3];   // (1000,)
    float*       out    = (float*)d_out;           // (B, 1000)

    const size_t rate_bytes = (size_t)B_ROWS * N_TOTAL * sizeof(_Float16); // 92.3 MB
    _Float16* rateH = (_Float16*)d_ws;
    _Float16* Wh    = (_Float16*)((char*)d_ws + rate_bytes);               // 1.44 MB

    sim_kernel<<<B_ROWS / 8, 256, 0, stream>>>(x, phase0, rateH);
    convw_kernel<<<(N_PAD * N_TOTAL) / 256, 256, 0, stream>>>(W, Wh);

    const size_t lds_bytes = (size_t)64 * N_TOTAL * sizeof(_Float16);      // 90112 B
    gemm_kernel<<<dim3(B_ROWS / 256, N_PAD / 64), 256, lds_bytes, stream>>>(
        rateH, Wh, bias, out);
}
